// EDN_M_91130616087315
// MI455X (gfx1250) — compile-verified, matrix operand order checked by reading, not run on hardware
//
#include <hip/hip_runtime.h>
#include <hip/hip_bf16.h>
#include <math.h>

typedef __attribute__((ext_vector_type(16))) _Float16 v16h;
typedef __attribute__((ext_vector_type(8)))  _Float16 v8h;
typedef __attribute__((ext_vector_type(4)))  _Float16 v4h;
typedef __attribute__((ext_vector_type(8)))  float    v8f;
typedef __attribute__((ext_vector_type(4)))  float    v4f;

// ---------------------------------------------------------------------------
// WMMA helpers (CDNA5 gfx1250, wave32).  D = A(16x32 f16) * B(32x16 f16) + C
// ---------------------------------------------------------------------------
__device__ __forceinline__ v8f wmma16(v16h a, v16h b, v8f c) {
  return __builtin_amdgcn_wmma_f32_16x16x32_f16(false, a, false, b, (short)0, c,
                                                false, false);
}

// Fragment from a row-major f16 matrix (row stride ldk, 16B-aligned rows).
// ISA layout: lane L -> row = L%16, K = kb + {0..7, 16..23}, kb = (L>=16)?8:0.
// Both K runs are contiguous -> two 128-bit loads per lane.
__device__ __forceinline__ v16h frag_rm(const _Float16* A, int ldk, int kofs, int lane) {
  int m  = lane & 15;
  int kb = (lane & 16) ? 8 : 0;
  const _Float16* p = A + m * ldk + kofs + kb;
  v8h lo = *(const v8h*)(p);
  v8h hi = *(const v8h*)(p + 16);
  v16h r;
#pragma unroll
  for (int e = 0; e < 8; ++e) { r[e] = lo[e]; r[e + 8] = hi[e]; }
  return r;
}

// C store: lane L -> N = L%16, rows M = ((L>=16)?8:0)+r
__device__ __forceinline__ void store_relu_h(_Float16* D, int ldn, int nofs, int Nact,
                                             v8f c, const float* bias, int lane) {
  int n = nofs + (lane & 15);
  int mrow = (lane & 16) ? 8 : 0;
  if (n < Nact) {
    float bv = bias[n];
#pragma unroll
    for (int r = 0; r < 8; ++r)
      D[(mrow + r) * ldn + n] = (_Float16)fmaxf(c[r] + bv, 0.f);
  }
}

__device__ __forceinline__ void store_lin_f(float* D, int ldn, int nofs, int Nact,
                                            v8f c, const float* bias, int lane) {
  int n = nofs + (lane & 15);
  int mrow = (lane & 16) ? 8 : 0;
  if (n < Nact) {
    float bv = bias[n];
#pragma unroll
    for (int r = 0; r < 8; ++r)
      D[(mrow + r) * ldn + n] = c[r] + bv;
  }
}

// ---------------------------------------------------------------------------
// Weight packing: fp32 -> f16, transposed to N-major (K contiguous), padded.
// ---------------------------------------------------------------------------
__global__ __launch_bounds__(256) void k_pack_lin(const float* __restrict__ W,
                                                  int K, int N, int Kpad, int Npad,
                                                  _Float16* __restrict__ Wt) {
  int i = blockIdx.x * 256 + threadIdx.x;
  if (i < Npad * Kpad) {
    int n = i / Kpad, k = i - n * Kpad;
    float v = (k < K && n < N) ? W[k * N + n] : 0.f;
    Wt[i] = (_Float16)v;
  }
}
// conv weights are already N-major (OIHW flat = o*K + k); just pad K.
__global__ __launch_bounds__(256) void k_pack_conv(const float* __restrict__ W,
                                                   int K, int Kpad, int N,
                                                   _Float16* __restrict__ Wt) {
  int i = blockIdx.x * 256 + threadIdx.x;
  if (i < N * Kpad) {
    int n = i / Kpad, k = i - n * Kpad;
    float v = (k < K) ? W[n * K + k] : 0.f;
    Wt[i] = (_Float16)v;
  }
}

// ---------------------------------------------------------------------------
// Kernel 1: feature prep + traj/vehs 3-layer MLPs.  1 block = 16 batch rows.
// waves 0-3 -> traj path, waves 4-7 -> vehs path.  Writes Xall cols [0,128).
// ---------------------------------------------------------------------------
__global__ __launch_bounds__(256) void k_prep_mlp(
    const float* __restrict__ obs, const float* __restrict__ sr_,
    const float* __restrict__ ps_,
    const _Float16* tW1p, const float* tb1, const _Float16* tW2p, const float* tb2,
    const _Float16* tW3p, const float* tb3,
    const _Float16* vW1p, const float* vb1, const _Float16* vW2p, const float* vb2,
    const _Float16* vW3p, const float* vb3,
    _Float16* __restrict__ Xall, int B) {
  __shared__ __align__(16) _Float16 X[2][16][32];
  __shared__ __align__(16) _Float16 Y[2][16][32];
  int tid = threadIdx.x, lane = tid & 31, wave = tid >> 5;
  int base = blockIdx.x * 16;

  if (tid < 16) {
    int b = base + tid;
    if (b < B) {
      float sr = sr_[b], ps = ps_[b], s = sr / ps;
      const float* ob = obs + (size_t)b * 8 * 16 * 2;
#pragma unroll
      for (int i = 0; i < 7; ++i)
        for (int d = 0; d < 2; ++d) {
          float dv = (ob[((i + 1) * 16) * 2 + d] - ob[(i * 16) * 2 + d]) * s;
          X[0][tid][i * 2 + d] = (_Float16)dv;
        }
      for (int k = 14; k < 32; ++k) X[0][tid][k] = (_Float16)0.f;

      float ex = ob[(7 * 16 + 0) * 2 + 0];
      float cx[16], cy[16], dist[16];
      for (int j = 0; j < 16; ++j) {
        float xj = ob[(7 * 16 + j) * 2 + 0];
        float yj = ob[(7 * 16 + j) * 2 + 1];
        float a = xj, c = yj;
        if (!(ex >= xj)) { a = -1000.f; c = -1000.f; }
        if (a != a) a = -1000.f;
        if (c != c) c = -1000.f;
        cx[j] = a; cy[j] = c;
      }
      for (int j = 0; j < 16; ++j) {
        float dx = cx[j] - cx[0], dy = cy[j] - cy[0];
        dist[j] = dx * dx + dy * dy;
      }
      int sel[4]; bool used[16];
      for (int j = 0; j < 16; ++j) used[j] = false;
      for (int t = 0; t < 4; ++t) {
        int bi = -1; float bd = __builtin_inff();
        for (int j = 0; j < 16; ++j)
          if (!used[j] && dist[j] < bd) { bd = dist[j]; bi = j; }
        if (bi < 0) { for (int j = 0; j < 16; ++j) if (!used[j]) { bi = j; break; } }
        used[bi] = true; sel[t] = bi;
      }
      for (int kk = 0; kk < 3; ++kk) {
        int j = sel[kk + 1];
        float cd = dist[j];
        float rel[6]; float rsum = 0.f;
        for (int t = 0; t < 3; ++t)
          for (int d = 0; d < 2; ++d) {
            float r = ob[(7 * 16 + 0) * 2 + d] - ob[((5 + t) * 16 + j) * 2 + d];
            rel[t * 2 + d] = r; rsum += r;
          }
        bool bad = (cd > 600000.f) || (rsum != rsum);
        for (int t = 0; t < 3; ++t)
          for (int d = 0; d < 2; ++d) {
            float v = bad ? 2000.f : rel[t * 2 + d];
            X[1][tid][t * 6 + kk * 2 + d] = (_Float16)(v * s);
          }
      }
      for (int k = 18; k < 32; ++k) X[1][tid][k] = (_Float16)0.f;
    } else {
      for (int k = 0; k < 32; ++k) { X[0][tid][k] = (_Float16)0.f; X[1][tid][k] = (_Float16)0.f; }
    }
  }
  __syncthreads();

  int grp = wave >> 2;      // 0=traj 1=vehs
  int wl  = wave & 3;
  // Layer 1: X -> Y, K=32(pad), N=32  (waves 0,1 / 4,5)
  if (wl < 2) {
    const _Float16* Wt = grp ? vW1p : tW1p;
    const float* bi = grp ? vb1 : tb1;
    v16h a  = frag_rm(&X[grp][0][0], 32, 0, lane);
    v16h bf = frag_rm(Wt + wl * 16 * 32, 32, 0, lane);
    v8f c = {};
    c = wmma16(a, bf, c);
    store_relu_h(&Y[grp][0][0], 32, wl * 16, 32, c, bi, lane);
  }
  __syncthreads();
  // Layer 2: Y -> X, K=32, N=32
  if (wl < 2) {
    const _Float16* Wt = grp ? vW2p : tW2p;
    const float* bi = grp ? vb2 : tb2;
    v16h a  = frag_rm(&Y[grp][0][0], 32, 0, lane);
    v16h bf = frag_rm(Wt + wl * 16 * 32, 32, 0, lane);
    v8f c = {};
    c = wmma16(a, bf, c);
    store_relu_h(&X[grp][0][0], 32, wl * 16, 32, c, bi, lane);
  }
  __syncthreads();
  // Layer 3 (linear): X -> Xall[:, grp*64 + n], K=32, N=64 (all 8 waves)
  {
    const _Float16* Wt = grp ? vW3p : tW3p;
    const float* bi = grp ? vb3 : tb3;
    v16h a  = frag_rm(&X[grp][0][0], 32, 0, lane);
    v16h bf = frag_rm(Wt + wl * 16 * 32, 32, 0, lane);
    v8f c = {};
    c = wmma16(a, bf, c);
    int n = wl * 16 + (lane & 15);
    int mrow = (lane & 16) ? 8 : 0;
    float bv = bi[n];
#pragma unroll
    for (int r = 0; r < 8; ++r) {
      int row = base + mrow + r;
      if (row < B)
        Xall[(size_t)row * 256 + grp * 64 + n] = (_Float16)(c[r] + bv);
    }
  }
}

// ---------------------------------------------------------------------------
// Kernel 2: CNN (3x3 s2 3->16 relu; 3x3 s2 16->32 relu; mean; fc 32->128).
// 1 block per element, implicit-GEMM WMMA. SAME pad = (0,1). Writes cols [128,256).
// ---------------------------------------------------------------------------
__global__ __launch_bounds__(256) void k_cnn(
    const float* __restrict__ scene,
    const _Float16* cW1p, const float* cb1, const _Float16* cW2p, const float* cb2,
    const float* cWfc, const float* cbfc,
    _Float16* __restrict__ Xall, int B) {
  __shared__ __align__(16) _Float16 S[3 * 32 * 32];   // scene f16 [c][y][x]
  __shared__ __align__(16) _Float16 C1[256 * 16];     // conv1 out [pixel][ch]
  __shared__ float partial[8 * 32];
  __shared__ float meanv[32];
  int tid = threadIdx.x, lane = tid & 31, wave = tid >> 5;
  int b = blockIdx.x;
  const float* sc = scene + (size_t)b * 3072;
  __builtin_prefetch(sc, 0, 0);                       // global_prefetch_b8
  for (int i = tid * 4; i < 3072; i += 1024) {
    v4f f = *(const v4f*)(sc + i);
    v4h h;
#pragma unroll
    for (int e = 0; e < 4; ++e) h[e] = (_Float16)f[e];
    *(v4h*)(S + i) = h;
  }
  __syncthreads();

  // conv1: 256 out pixels x 16 ch, K = 27 (padded 32), 16 pixel tiles
  for (int pt = wave; pt < 16; pt += 8) {
    int m = lane & 15;
    int p = pt * 16 + m, oy = p >> 4, ox = p & 15;
    int kb = (lane & 16) ? 8 : 0;
    v16h a;
#pragma unroll
    for (int e = 0; e < 16; ++e) {
      int k = kb + e + ((e >= 8) ? 8 : 0);
      _Float16 v = (_Float16)0.f;
      if (k < 27) {
        int c = k / 9, r = k % 9, ky = r / 3, kx = r % 3;
        int iy = 2 * oy + ky, ix = 2 * ox + kx;
        if (iy < 32 && ix < 32) v = S[c * 1024 + iy * 32 + ix];
      }
      a[e] = v;
    }
    v16h bf = frag_rm(cW1p, 32, 0, lane);
    v8f c = {};
    c = wmma16(a, bf, c);
    int n = lane & 15, mrow = (lane & 16) ? 8 : 0;
    float bv = cb1[n];
#pragma unroll
    for (int r = 0; r < 8; ++r)
      C1[(pt * 16 + mrow + r) * 16 + n] = (_Float16)fmaxf(c[r] + bv, 0.f);
  }
  __syncthreads();

  // conv2: 64 out pixels x 32 ch, K = 144 (padded 160, 5 chunks of 32)
  {
    int ptile = wave & 3, ct = wave >> 2;
    int m = lane & 15;
    int p = ptile * 16 + m, oy = p >> 3, ox = p & 7;
    int kb = (lane & 16) ? 8 : 0;
    v8f acc = {};
    for (int kc = 0; kc < 5; ++kc) {
      v16h a;
#pragma unroll
      for (int e = 0; e < 16; ++e) {
        int k = kc * 32 + kb + e + ((e >= 8) ? 8 : 0);
        _Float16 v = (_Float16)0.f;
        if (k < 144) {
          int ic = k / 9, r = k % 9, ky = r / 3, kx = r % 3;
          int iy = 2 * oy + ky, ix = 2 * ox + kx;
          if (iy < 16 && ix < 16) v = C1[(iy * 16 + ix) * 16 + ic];
        }
        a[e] = v;
      }
      v16h bf = frag_rm(cW2p + (ct * 16) * 160, 160, kc * 32, lane);
      acc = wmma16(a, bf, acc);
    }
    int n = ct * 16 + (lane & 15);
    float bv = cb2[n], s = 0.f;
#pragma unroll
    for (int r = 0; r < 8; ++r) s += fmaxf(acc[r] + bv, 0.f);
    partial[wave * 32 + lane] = s;
  }
  __syncthreads();
  if (tid < 32) {
    int ct = tid >> 4, nl = tid & 15;
    float s = 0.f;
    for (int pt = 0; pt < 4; ++pt) {
      int w = ct * 4 + pt;
      s += partial[w * 32 + nl] + partial[w * 32 + nl + 16];
    }
    meanv[tid] = s * (1.f / 64.f);
  }
  __syncthreads();
  if (tid < 128) {
    float f = cbfc[tid];
    for (int k = 0; k < 32; ++k) f += meanv[k] * cWfc[k * 128 + tid];
    Xall[(size_t)b * 256 + 128 + tid] = (_Float16)f;
  }
}

// ---------------------------------------------------------------------------
// Kernel 3: decoder + mean/var/prob heads + epilogue.  1 block = 16 rows.
// ---------------------------------------------------------------------------
__global__ __launch_bounds__(256) void k_heads(
    const float* __restrict__ obs, const float* __restrict__ sr_,
    const float* __restrict__ ps_,
    const _Float16* __restrict__ Xall,
    const _Float16* decWp, const float* decb,
    const _Float16* mW1p, const float* mb1, const _Float16* mW2p, const float* mb2,
    const _Float16* vW1p, const float* vb1, const _Float16* vW2p, const float* vb2,
    const _Float16* pW1p, const float* pb1, const _Float16* pW2p, const float* pb2,
    float* __restrict__ out, int B) {
  __shared__ __align__(16) _Float16 X[16 * 256];
  __shared__ __align__(16) _Float16 F[16 * 128];
  __shared__ __align__(16) _Float16 H[16 * 128];
  __shared__ float MO[16 * 48];
  __shared__ float VO[16 * 72];
  __shared__ float PO[16 * 2];
  int tid = threadIdx.x, lane = tid & 31, wave = tid >> 5;
  int base = blockIdx.x * 16;

  // stage 16x256 f16 rows with 128-bit copies
  for (int i = tid; i < 512; i += 256) {
    int row = i >> 5;
    int cof = (i & 31) * 8;
    v8h v = {};
    int b = base + row;
    if (b < B) v = *(const v8h*)(Xall + (size_t)b * 256 + cof);
    *(v8h*)(X + row * 256 + cof) = v;
  }
  __syncthreads();

  // decoder: 256 -> 128, relu
  {
    v8f acc = {};
    for (int kc = 0; kc < 8; ++kc) {
      v16h a  = frag_rm(X, 256, kc * 32, lane);
      v16h bf = frag_rm(decWp + (wave * 16) * 256, 256, kc * 32, lane);
      acc = wmma16(a, bf, acc);
    }
    store_relu_h(F, 128, wave * 16, 128, acc, decb, lane);
  }
  __syncthreads();
  // mean hidden 128->128 relu
  {
    v8f acc = {};
    for (int kc = 0; kc < 4; ++kc) {
      v16h a  = frag_rm(F, 128, kc * 32, lane);
      v16h bf = frag_rm(mW1p + (wave * 16) * 128, 128, kc * 32, lane);
      acc = wmma16(a, bf, acc);
    }
    store_relu_h(H, 128, wave * 16, 128, acc, mb1, lane);
  }
  __syncthreads();
  // mean out 128->48
  if (wave < 3) {
    v8f acc = {};
    for (int kc = 0; kc < 4; ++kc) {
      v16h a  = frag_rm(H, 128, kc * 32, lane);
      v16h bf = frag_rm(mW2p + (wave * 16) * 128, 128, kc * 32, lane);
      acc = wmma16(a, bf, acc);
    }
    store_lin_f(MO, 48, wave * 16, 48, acc, mb2, lane);
  }
  __syncthreads();
  // var hidden 128->128 relu (reuse H)
  {
    v8f acc = {};
    for (int kc = 0; kc < 4; ++kc) {
      v16h a  = frag_rm(F, 128, kc * 32, lane);
      v16h bf = frag_rm(vW1p + (wave * 16) * 128, 128, kc * 32, lane);
      acc = wmma16(a, bf, acc);
    }
    store_relu_h(H, 128, wave * 16, 128, acc, vb1, lane);
  }
  __syncthreads();
  // var out 128->72 (packed Npad=80, so tile 4 reads zero rows)
  if (wave < 5) {
    v8f acc = {};
    for (int kc = 0; kc < 4; ++kc) {
      v16h a  = frag_rm(H, 128, kc * 32, lane);
      v16h bf = frag_rm(vW2p + (wave * 16) * 128, 128, kc * 32, lane);
      acc = wmma16(a, bf, acc);
    }
    store_lin_f(VO, 72, wave * 16, 72, acc, vb2, lane);
  }
  __syncthreads();
  // prob hidden 128->32 relu (reuse H as 16x32)
  if (wave < 2) {
    v8f acc = {};
    for (int kc = 0; kc < 4; ++kc) {
      v16h a  = frag_rm(F, 128, kc * 32, lane);
      v16h bf = frag_rm(pW1p + (wave * 16) * 128, 128, kc * 32, lane);
      acc = wmma16(a, bf, acc);
    }
    store_relu_h(H, 32, wave * 16, 32, acc, pb1, lane);
  }
  __syncthreads();
  // prob out 32->2 (packed Npad=16)
  if (wave == 0) {
    v8f acc = {};
    for (int kc = 0; kc < 2; ++kc) {
      v16h a  = frag_rm(H, 32, kc * 32, lane);
      v16h bf = frag_rm(pW2p, 32, kc * 32, lane);
      acc = wmma16(a, bf, acc);
    }
    store_lin_f(PO, 2, 0, 2, acc, pb2, lane);
  }
  __syncthreads();

  // epilogue: one thread per (row, mode)
  if (tid < 32) {
    int row = tid >> 1, m = tid & 1;
    int b = base + row;
    if (b < B) {
      float sr = sr_[b], ps = ps_[b], scale = ps / sr;
      const float* ob = obs + (size_t)b * 256;
      float cumx = ob[(7 * 16 + 0) * 2 + 0];
      float cumy = ob[(7 * 16 + 0) * 2 + 1];
      size_t opp = (size_t)B * 120;   // prediction_pos
      size_t opr = (size_t)B * 240;   // prob
      size_t osp = (size_t)B * 242;   // speed
      for (int j = 0; j < 12; ++j) {
        float mx = MO[row * 48 + (m * 12 + j) * 2 + 0];
        float my = MO[row * 48 + (m * 12 + j) * 2 + 1];
        float v0 = VO[row * 72 + (m * 12 + j) * 3 + 0];
        float v1 = VO[row * 72 + (m * 12 + j) * 3 + 1];
        float v2 = VO[row * 72 + (m * 12 + j) * 3 + 2];
        float e0 = expf(v0), e1 = expf(v1), t2 = tanhf(v2);
        float svx = mx * scale, svy = my * scale;
        float spd = sqrtf(mx * mx + my * my) * scale;
        size_t ib = ((size_t)(b * 2 + m) * 12 + j) * 5;
        out[ib + 0] = svx;  out[ib + 1] = svy;
        out[ib + 2] = e0 * scale; out[ib + 3] = e1 * scale; out[ib + 4] = t2 * scale;
        cumx += svx; cumy += svy;
        out[opp + ib + 0] = cumx; out[opp + ib + 1] = cumy;
        out[opp + ib + 2] = e0;   out[opp + ib + 3] = e1;   out[opp + ib + 4] = t2;
        out[osp + (size_t)(b * 2 + m) * 12 + j] = spd;
      }
      out[opr + (size_t)b * 2 + m] = PO[row * 2 + m];
    }
  }
}

// ---------------------------------------------------------------------------
extern "C" void kernel_launch(void* const* d_in, const int* in_sizes, int n_in,
                              void* d_out, int out_size, void* d_ws, size_t ws_size,
                              hipStream_t stream) {
  const float* obs   = (const float*)d_in[0];
  const float* scene = (const float*)d_in[1];
  const float* sr    = (const float*)d_in[2];
  const float* ps    = (const float*)d_in[3];
  const float* tW1 = (const float*)d_in[4];   const float* tb1 = (const float*)d_in[5];
  const float* tW2 = (const float*)d_in[6];   const float* tb2 = (const float*)d_in[7];
  const float* tW3 = (const float*)d_in[8];   const float* tb3 = (const float*)d_in[9];
  const float* vW1 = (const float*)d_in[10];  const float* vb1 = (const float*)d_in[11];
  const float* vW2 = (const float*)d_in[12];  const float* vb2 = (const float*)d_in[13];
  const float* vW3 = (const float*)d_in[14];  const float* vb3 = (const float*)d_in[15];
  const float* decW = (const float*)d_in[16]; const float* decb = (const float*)d_in[17];
  const float* mW1 = (const float*)d_in[18];  const float* mb1 = (const float*)d_in[19];
  const float* mW2 = (const float*)d_in[20];  const float* mb2 = (const float*)d_in[21];
  const float* vaW1 = (const float*)d_in[22]; const float* vab1 = (const float*)d_in[23];
  const float* vaW2 = (const float*)d_in[24]; const float* vab2 = (const float*)d_in[25];
  const float* pW1 = (const float*)d_in[26];  const float* pb1 = (const float*)d_in[27];
  const float* pW2 = (const float*)d_in[28];  const float* pb2 = (const float*)d_in[29];
  const float* cW1 = (const float*)d_in[30];  const float* cb1 = (const float*)d_in[31];
  const float* cW2 = (const float*)d_in[32];  const float* cb2 = (const float*)d_in[33];
  const float* cWfc = (const float*)d_in[34]; const float* cbfc = (const float*)d_in[35];

  int B = in_sizes[2];  // sample_rate length

  // ws layout: Xall (B x 256 f16), then packed weights
  _Float16* Xall = (_Float16*)d_ws;
  _Float16* wp = Xall + (size_t)B * 256;
  _Float16 *tW1p, *tW2p, *tW3p, *vW1p, *vW2p, *vW3p, *decWp;
  _Float16 *mW1p, *mW2p, *vaW1p, *vaW2p, *pW1p, *pW2p, *cW1p, *cW2p;
  struct PackL { const float* W; int K, N, Kp, Np; _Float16** dst; };
  PackL packs[] = {
      {tW1, 14, 32, 32, 32, &tW1p},   {tW2, 32, 32, 32, 32, &tW2p},
      {tW3, 32, 64, 32, 64, &tW3p},   {vW1, 18, 32, 32, 32, &vW1p},
      {vW2, 32, 32, 32, 32, &vW2p},   {vW3, 32, 64, 32, 64, &vW3p},
      {decW, 256, 128, 256, 128, &decWp},
      {mW1, 128, 128, 128, 128, &mW1p}, {mW2, 128, 48, 128, 48, &mW2p},
      {vaW1, 128, 128, 128, 128, &vaW1p}, {vaW2, 128, 72, 128, 80, &vaW2p},
      {pW1, 128, 32, 128, 32, &pW1p}, {pW2, 32, 2, 32, 16, &pW2p},
  };
  for (auto& pk : packs) {
    *pk.dst = wp;
    int sz = pk.Kp * pk.Np;
    k_pack_lin<<<(sz + 255) / 256, 256, 0, stream>>>(pk.W, pk.K, pk.N, pk.Kp, pk.Np, wp);
    wp += sz;
  }
  cW1p = wp;
  k_pack_conv<<<(16 * 32 + 255) / 256, 256, 0, stream>>>(cW1, 27, 32, 16, cW1p);
  wp += 16 * 32;
  cW2p = wp;
  k_pack_conv<<<(32 * 160 + 255) / 256, 256, 0, stream>>>(cW2, 144, 160, 32, cW2p);
  wp += 32 * 160;

  int nb = (B + 15) / 16;
  k_prep_mlp<<<nb, 256, 0, stream>>>(obs, sr, ps,
      tW1p, tb1, tW2p, tb2, tW3p, tb3, vW1p, vb1, vW2p, vb2, vW3p, vb3,
      Xall, B);
  k_cnn<<<B, 256, 0, stream>>>(scene, cW1p, cb1, cW2p, cb2, cWfc, cbfc, Xall, B);
  k_heads<<<nb, 256, 0, stream>>>(obs, sr, ps, Xall,
      decWp, decb, mW1p, mb1, mW2p, mb2, vaW1p, vab1, vaW2p, vab2,
      pW1p, pb1, pW2p, pb2, (float*)d_out, B);
}
